// Netsum_10831907520693
// MI455X (gfx1250) — matrix operand order repaired and checked by reading, not compiled, not run on hardware
//
#include <hip/hip_runtime.h>
#include <hip/hip_bf16.h>

typedef __bf16 bf16_t;
typedef __attribute__((ext_vector_type(16))) __bf16 v16bf;
typedef __attribute__((ext_vector_type(8)))  __bf16 v8bf;
typedef __attribute__((ext_vector_type(4)))  __bf16 v4bf;
typedef __attribute__((ext_vector_type(8)))  float  v8f;
typedef __attribute__((ext_vector_type(4)))  float  v4f;

#define B_   65536
#define D_   256
#define H_   512
#define C_   10
#define P_   10
#define HP_  128
#define TM   128          // batch rows per workgroup
#define XPITCH 264        // bf16 elems per row in LDS x tile (skewed)
#define HPITCH 136        // bf16 elems per row in LDS hidden tile (skewed)

// ---------------------------------------------------------------------------
// Pack a KxN f32 matrix into the WMMA bf16 B-matrix tile layout:
//   tile (nt,kt) -> tileIdx = nt*Kt + kt, 1024 bytes per tile:
//   [lane 0..31][16 bf16], element (2v+h) = B[kt*32 + 2v + h + (lane>=16?16:0)]
//                                            [nt*16 + (lane&15)]  (0 if col>=N)
// blockIdx.y = p (for stacked per-patch weights)
// ---------------------------------------------------------------------------
__global__ void pack_b_kernel(const float* __restrict__ src, bf16_t* __restrict__ dst,
                              int K, int N, int Ntp,
                              size_t srcStride, size_t dstStride) {
    const int p = blockIdx.y;
    src += (size_t)p * srcStride;
    dst += (size_t)p * dstStride;
    const int Kt = K >> 5;
    const int total = Ntp * Kt * 32;
    int idx = blockIdx.x * blockDim.x + threadIdx.x;
    if (idx >= total) return;
    const int lane = idx & 31;
    const int tile = idx >> 5;
    const int kt = tile % Kt;
    const int nt = tile / Kt;
    const int koff = kt * 32 + ((lane >= 16) ? 16 : 0);
    const int n = nt * 16 + (lane & 15);
    bf16_t* out = dst + (size_t)tile * 512 + lane * 16;
#pragma unroll
    for (int v = 0; v < 8; ++v) {
#pragma unroll
        for (int h = 0; h < 2; ++h) {
            const int k = koff + 2 * v + h;
            const float val = (n < N) ? src[(size_t)k * N + n] : 0.0f;
            out[v * 2 + h] = (bf16_t)val;
        }
    }
}

// A-matrix fragment (16x32 bf16) from row-major bf16 LDS tile.
// lane<16: row m0+lane, K {k0..k0+15}
// lane>=16: row m0+lane-16, K {k0+16..k0+31}
__device__ __forceinline__ v16bf load_A_lds(const bf16_t* base, int pitch,
                                            int m0, int k0, int lane) {
    const bf16_t* p = base + (size_t)(m0 + (lane & 15)) * pitch + k0 + ((lane >> 4) << 4);
    v8bf lo = *(const v8bf*)p;
    v8bf hi = *(const v8bf*)(p + 8);
    return __builtin_shufflevector(lo, hi, 0, 1, 2, 3, 4, 5, 6, 7,
                                           8, 9, 10, 11, 12, 13, 14, 15);
}

// B-matrix fragment from pre-packed global tiles (32 contiguous bytes per lane).
__device__ __forceinline__ v16bf load_B_glb(const bf16_t* packed, int tile, int lane) {
    return *(const v16bf*)(packed + (size_t)tile * 512 + lane * 16);
}

__device__ __forceinline__ v8f wmma_bf16(v16bf a, v16bf b, v8f c) {
    return __builtin_amdgcn_wmma_f32_16x16x32_bf16(false, a, false, b,
                                                   (short)0, c, false, false);
}

// ---------------------------------------------------------------------------
// Fused: target net + top-2 bitmap + 10 gated patch nets, one x read.
// 8 waves; wave w owns output row-tile [w*16, w*16+16).
// ---------------------------------------------------------------------------
__global__ __launch_bounds__(256, 1)
void fused_netsum_kernel(const float* __restrict__ x,
                         const float* __restrict__ bt1, const float* __restrict__ bt2,
                         const float* __restrict__ bp1, const float* __restrict__ bp2,
                         const int*   __restrict__ rdd,
                         const bf16_t* __restrict__ Wt1p, const bf16_t* __restrict__ Wt2p,
                         const bf16_t* __restrict__ Wp1p, const bf16_t* __restrict__ Wp2p,
                         float* __restrict__ outg) {
    __shared__ __align__(16) bf16_t xs[TM * XPITCH];   // x tile, bf16
    __shared__ __align__(16) bf16_t hs[TM * HPITCH];   // hidden chunk, bf16
    __shared__ float outs[TM * 16];                    // target logits (padded)
    __shared__ float bmap[TM * 16];                    // bitmap as float

    const int tid = threadIdx.x;
    const int wave = tid >> 5;
    const int lane = tid & 31;
    const int col = lane & 15;              // N index within a 16-wide tile
    const int rsel = (lane >> 4) << 3;      // C/D row offset: 0 or 8
    const int rowBase = blockIdx.x * TM;

    // ---- stage x: f32 -> bf16 into LDS (coalesced b128 loads) ----
    const v4f* x4 = (const v4f*)(x + (size_t)rowBase * D_);
    for (int i = tid; i < TM * (D_ / 4); i += 256) {
        const int r = i >> 6;               // D_/4 == 64
        const int c4 = i & 63;
        v4f v = x4[(size_t)r * 64 + c4];
        v4bf b;
        b.x = (bf16_t)v.x; b.y = (bf16_t)v.y; b.z = (bf16_t)v.z; b.w = (bf16_t)v.w;
        *(v4bf*)&xs[r * XPITCH + c4 * 4] = b;
    }
    __syncthreads();

    // ---- target net, H processed in 4 chunks of 128 ----
    v8f cout = {0.f, 0.f, 0.f, 0.f, 0.f, 0.f, 0.f, 0.f};
    for (int chunk = 0; chunk < 4; ++chunk) {
        const int nt = chunk * 8 + wave;            // hidden col-tile 0..31
        const float b1 = bt1[nt * 16 + col];
        for (int rt = 0; rt < 8; ++rt) {
            v8f c = {0.f, 0.f, 0.f, 0.f, 0.f, 0.f, 0.f, 0.f};
#pragma unroll
            for (int kt = 0; kt < 8; ++kt) {
                v16bf a = load_A_lds(xs, XPITCH, rt * 16, kt * 32, lane);
                v16bf b = load_B_glb(Wt1p, nt * 8 + kt, lane);
                c = wmma_bf16(a, b, c);
            }
            const int hrow0 = rt * 16 + rsel;
#pragma unroll
            for (int j = 0; j < 8; ++j) {
                float v = c[j] + b1;
                v = v > 0.f ? v : 0.f;              // relu
                hs[(hrow0 + j) * HPITCH + wave * 16 + col] = (bf16_t)v;
            }
        }
        __syncthreads();
        // layer-2 partial over this chunk (wave's row tile)
#pragma unroll
        for (int kt = 0; kt < 4; ++kt) {
            v16bf a = load_A_lds(hs, HPITCH, wave * 16, kt * 32, lane);
            v16bf b = load_B_glb(Wt2p, chunk * 4 + kt, lane);
            cout = wmma_bf16(a, b, cout);
        }
        __syncthreads();                            // hs reused next chunk
    }
    const float b2 = (col < C_) ? bt2[col] : 0.0f;
#pragma unroll
    for (int j = 0; j < 8; ++j) {
        cout[j] += b2;
        outs[(wave * 16 + rsel + j) * 16 + col] = cout[j];
    }
    __syncthreads();

    // ---- top-2 + bitmap (one thread per batch row) ----
    if (tid < TM) {
        const float* o = &outs[tid * 16];
        int i0 = 0; float v0 = o[0];
        for (int c = 1; c < C_; ++c) { float v = o[c]; if (v > v0) { v0 = v; i0 = c; } }
        int i1 = -1; float v1 = -3.4e38f;
        for (int c = 0; c < C_; ++c) {
            if (c == i0) continue;
            float v = o[c]; if (v > v1) { v1 = v; i1 = c; }
        }
        bool any = false;
        float fb[P_];
#pragma unroll
        for (int p = 0; p < P_; ++p) {
            const int r0 = rdd[2 * p], r1 = rdd[2 * p + 1];
            const bool m = (i0 == r0) && (i1 == r1);
            any = any || m;
            bmap[tid * 16 + p] = m ? 1.0f : 0.0f;
            fb[p] = (i0 == r0) ? 1.0f : 0.0f;
        }
        if (!any) {
#pragma unroll
            for (int p = 0; p < P_; ++p) bmap[tid * 16 + p] = fb[p];
        }
    }
    __syncthreads();

    // ---- 10 patch nets, gated accumulate into cout ----
    for (int p = 0; p < P_; ++p) {
        const bf16_t* W1 = Wp1p + (size_t)p * (64 * 512);
        const bf16_t* W2 = Wp2p + (size_t)p * (4 * 512);
        const float pb1 = bp1[p * HP_ + wave * 16 + col];
        for (int rt = 0; rt < 8; ++rt) {
            v8f c = {0.f, 0.f, 0.f, 0.f, 0.f, 0.f, 0.f, 0.f};
#pragma unroll
            for (int kt = 0; kt < 8; ++kt) {
                v16bf a = load_A_lds(xs, XPITCH, rt * 16, kt * 32, lane);
                v16bf b = load_B_glb(W1, wave * 8 + kt, lane);
                c = wmma_bf16(a, b, c);
            }
            const int hrow0 = rt * 16 + rsel;
#pragma unroll
            for (int j = 0; j < 8; ++j) {
                float v = c[j] + pb1;
                v = v > 0.f ? v : 0.f;
                hs[(hrow0 + j) * HPITCH + wave * 16 + col] = (bf16_t)v;
            }
        }
        __syncthreads();
        v8f cp = {0.f, 0.f, 0.f, 0.f, 0.f, 0.f, 0.f, 0.f};
#pragma unroll
        for (int kt = 0; kt < 4; ++kt) {
            v16bf a = load_A_lds(hs, HPITCH, wave * 16, kt * 32, lane);
            v16bf b = load_B_glb(W2, kt, lane);
            cp = wmma_bf16(a, b, cp);
        }
        const float pb2 = (col < C_) ? bp2[p * C_ + col] : 0.0f;
#pragma unroll
        for (int j = 0; j < 8; ++j) {
            const int row = wave * 16 + rsel + j;
            const float g = bmap[row * 16 + p];
            cout[j] += g * (cp[j] + pb2);
        }
        __syncthreads();                            // hs reused next patch
    }

    // ---- store final [TM, 10] ----
    if (col < C_) {
#pragma unroll
        for (int j = 0; j < 8; ++j) {
            const int row = rowBase + wave * 16 + rsel + j;
            outg[(size_t)row * C_ + col] = cout[j];
        }
    }
}

extern "C" void kernel_launch(void* const* d_in, const int* in_sizes, int n_in,
                              void* d_out, int out_size, void* d_ws, size_t ws_size,
                              hipStream_t stream) {
    const float* x   = (const float*)d_in[0];
    const float* Wt1 = (const float*)d_in[1];
    const float* bt1 = (const float*)d_in[2];
    const float* Wt2 = (const float*)d_in[3];
    const float* bt2 = (const float*)d_in[4];
    const float* Wp1 = (const float*)d_in[5];
    const float* bp1 = (const float*)d_in[6];
    const float* Wp2 = (const float*)d_in[7];
    const float* bp2 = (const float*)d_in[8];
    const int*   rdd = (const int*)d_in[9];

    // workspace layout (bf16 elements), all 32B-aligned
    bf16_t* Wt1p = (bf16_t*)d_ws;            // 32*8 tiles  = 131072 elems
    bf16_t* Wt2p = Wt1p + 131072;            // 1*16 tiles  =   8192 elems
    bf16_t* Wp1p = Wt2p + 8192;              // 10*8*8 tiles= 327680 elems
    bf16_t* Wp2p = Wp1p + 327680;            // 10*1*4 tiles=  20480 elems

    // pack weights into WMMA B-matrix tile layout (bf16)
    pack_b_kernel<<<dim3((32 * 8 * 32 + 255) / 256, 1), 256, 0, stream>>>(
        Wt1, Wt1p, D_, H_, 32, 0, 0);
    pack_b_kernel<<<dim3((1 * 16 * 32 + 255) / 256, 1), 256, 0, stream>>>(
        Wt2, Wt2p, H_, C_, 1, 0, 0);
    pack_b_kernel<<<dim3((8 * 8 * 32 + 255) / 256, P_), 256, 0, stream>>>(
        Wp1, Wp1p, D_, HP_, 8, (size_t)D_ * HP_, (size_t)64 * 512);
    pack_b_kernel<<<dim3((1 * 4 * 32 + 255) / 256, P_), 256, 0, stream>>>(
        Wp2, Wp2p, HP_, C_, 1, (size_t)HP_ * C_, (size_t)4 * 512);

    fused_netsum_kernel<<<dim3(B_ / TM), 256, 0, stream>>>(
        x, bt1, bt2, bp1, bp2, rdd, Wt1p, Wt2p, Wp1p, Wp2p, (float*)d_out);
}
